// SimpleModelWithPermutation_22591527977568
// MI455X (gfx1250) — matrix operand order repaired and checked by reading, not compiled, not run on hardware
//
#include <hip/hip_runtime.h>
#include <hip/hip_bf16.h>

typedef __attribute__((ext_vector_type(16))) __bf16 v16bf;
typedef __attribute__((ext_vector_type(8)))  float  v8f;
typedef __attribute__((ext_vector_type(4)))  int    v4i;

#define B_DIM 8
#define S_DIM 4096
#define H_DIM 2048
#define L_DIM 4

#define BM 128
#define BN 256
#define BK 32
#define LDT 40   // padded LDS row stride in bf16 (80B = 20 banks -> conflict-free frag loads)
#define NKSTEPS (H_DIM / BK)
#define ABUF_BYTES (BM * LDT * 2)
#define BBUF_BYTES (BN * LDT * 2)

// ---- CDNA5 async global->LDS path (guarded; falls back to sync staging) ----
#if __has_builtin(__builtin_amdgcn_global_load_async_to_lds_b128)
#define HAVE_ASYNC_LDS 1
typedef __attribute__((address_space(1))) v4i as1_v4i;   // global int4
typedef __attribute__((address_space(3))) v4i as3_v4i;   // LDS int4
#endif

__device__ __forceinline__ void async_cp16(const void* g, void* l) {
#ifdef HAVE_ASYNC_LDS
    __builtin_amdgcn_global_load_async_to_lds_b128((as1_v4i*)g, (as3_v4i*)l, 0, 0);
#else
    *(uint4*)l = *(const uint4*)g;
#endif
}

__device__ __forceinline__ void wait_async_done() {
#ifdef HAVE_ASYNC_LDS
#if __has_builtin(__builtin_amdgcn_s_wait_asynccnt)
    __builtin_amdgcn_s_wait_asynccnt(0);
#else
    asm volatile("s_wait_asynccnt 0x0" ::: "memory");
#endif
#endif
}

__device__ __forceinline__ void sched_fence() {
#if __has_builtin(__builtin_amdgcn_sched_barrier)
    __builtin_amdgcn_sched_barrier(0);
#endif
}

// ---------------- fp32 -> bf16 conversion (vectorized x8) ----------------
__global__ __launch_bounds__(256)
void cvt_f32_to_bf16(const float* __restrict__ in, __bf16* __restrict__ out, int n) {
    int i = (blockIdx.x * 256 + threadIdx.x) * 8;
    if (i + 8 <= n) {
        float4 a = *(const float4*)(in + i);
        float4 b = *(const float4*)(in + i + 4);
        __bf16 o[8] = {(__bf16)a.x, (__bf16)a.y, (__bf16)a.z, (__bf16)a.w,
                       (__bf16)b.x, (__bf16)b.y, (__bf16)b.z, (__bf16)b.w};
        *(uint4*)(out + i) = *(const uint4*)o;
    }
}

// ------- one fused layer: out = gather_perm(X) @ W^T + bias -------
// Block tile 128(M) x 256(N), K-step 32, 8 waves in a 2x4 grid, 64x64 per wave.
// Double-buffered LDS fed by GLOBAL_LOAD_ASYNC_TO_LDS_B128 (ASYNCcnt pipeline).
// K loop deliberately NOT unrolled: dynamic ping/pong keeps one copy of all
// pipeline state so total VGPRs stay below 256 (no s_set_vgpr_msb shuffling).
template <bool F32OUT>
__global__ __launch_bounds__(256)
void gemm_bf16_wmma(const __bf16* __restrict__ X,
                    const __bf16* __restrict__ W,
                    const float*  __restrict__ bias,
                    const int*    __restrict__ perm,
                    void* __restrict__ outp)
{
    __shared__ __attribute__((aligned(16))) __bf16 lA[2][BM * LDT];   // 2 x 10 KB
    __shared__ __attribute__((aligned(16))) __bf16 lB[2][BN * LDT];   // 2 x 20 KB

    const int tid   = threadIdx.x;
    const int bn    = blockIdx.x;          // N blocks: 8
    const int bm    = blockIdx.y;          // M blocks: 256
    const int wave  = tid >> 5;
    const int lane  = tid & 31;
    const int waveM = wave & 1;            // 2 wave-rows x 64 M
    const int waveN = wave >> 1;           // 4 wave-cols x 64 N

    // Cooperative staging as 32-bit BYTE offsets (keeps address state small).
    // 16B chunks: A has 512 (2/thread), B has 1024 (4/thread).
    int aGOff[2], aLOff[2];
    #pragma unroll
    for (int j = 0; j < 2; ++j) {
        int c   = tid + j * 256;
        int r   = c >> 2;
        int q   = (c & 3) * 16;            // byte offset within the 64B k-row chunk group
        int gr  = bm * BM + r;             // blocks never straddle batches (S/BM = 32)
        int b   = gr >> 12;                // S = 4096
        int s   = gr & (S_DIM - 1);
        int src = perm ? ((b << 12) | perm[s]) : gr;
        aGOff[j] = src * (H_DIM * 2) + q;  // < 134 MB, fits i32
        aLOff[j] = r * (LDT * 2) + q;
    }
    int bGOff[4], bLOff[4];
    #pragma unroll
    for (int j = 0; j < 4; ++j) {
        int c = tid + j * 256;
        int r = c >> 2;
        int q = (c & 3) * 16;
        bGOff[j] = (bn * BN + r) * (H_DIM * 2) + q;   // < 8 MB
        bLOff[j] = r * (LDT * 2) + q;
    }

    const char* Xb  = (const char*)X;
    const char* Wb  = (const char*)W;
    char*       lAb = (char*)&lA[0][0];
    char*       lBb = (char*)&lB[0][0];

    auto stage = [&](int kByte, int buf) {
        char* la = lAb + buf * ABUF_BYTES;
        char* lb = lBb + buf * BBUF_BYTES;
        #pragma unroll
        for (int j = 0; j < 2; ++j)
            async_cp16(Xb + aGOff[j] + kByte, la + aLOff[j]);
        #pragma unroll
        for (int j = 0; j < 4; ++j)
            async_cp16(Wb + bGOff[j] + kByte, lb + bLOff[j]);
    };

    v8f acc[4][4];
    #pragma unroll
    for (int tm = 0; tm < 4; ++tm)
        #pragma unroll
        for (int tn = 0; tn < 4; ++tn)
            acc[tm][tn] = (v8f)0.0f;

    const int lm = lane & 15;              // row/col index within a 16-tile
    const int kh = lane >> 4;              // K-half select (ISA 16-bit layouts)

    // Per-lane fragment base byte-offsets; tile steps become ds immediate offsets.
    const int aFrag = (waveM * 64 + lm) * (LDT * 2) + kh * 16;   // +tm*16*LDT*2, +0/+32
    const int bFrag = (waveN * 64 + lm) * (LDT * 2) + kh * 16;   // +tn*16*LDT*2, +0/+32

    stage(0, 0);                           // prologue fill of buffer 0

    #pragma clang loop unroll(disable)
    for (int kk = 0; kk < NKSTEPS; ++kk) {
        wait_async_done();                 // this wave's copies for tile kk are in LDS
        __syncthreads();                   // everyone's copies landed; prior reads done
        const int cur = kk & 1;
        if (kk + 1 < NKSTEPS)
            stage((kk + 1) * (BK * 2), cur ^ 1);   // overlap next tile with compute

        const char* a = lAb + cur * ABUF_BYTES + aFrag;
        const char* b = lBb + cur * BBUF_BYTES + bFrag;

        // B fragments stay resident (32 VGPRs, reused by all 4 A tiles).
        v16bf bfr[4];
        #pragma unroll
        for (int tn = 0; tn < 4; ++tn) {
            *((uint4*)&bfr[tn] + 0) = *(const uint4*)(b + tn * (16 * LDT * 2));
            *((uint4*)&bfr[tn] + 1) = *(const uint4*)(b + tn * (16 * LDT * 2) + 32);
        }
        // A fragments streamed one at a time (8 VGPRs live).
        #pragma unroll
        for (int tm = 0; tm < 4; ++tm) {
            v16bf afr;
            *((uint4*)&afr + 0) = *(const uint4*)(a + tm * (16 * LDT * 2));        // K=kh*8..+8
            *((uint4*)&afr + 1) = *(const uint4*)(a + tm * (16 * LDT * 2) + 32);   // K=16+kh*8..+8
            #pragma unroll
            for (int tn = 0; tn < 4; ++tn)
                acc[tm][tn] = __builtin_amdgcn_wmma_f32_16x16x32_bf16(
                    false, afr, false, bfr[tn],
                    (short)0, acc[tm][tn], false, false);
            sched_fence();                 // keep afr live range short (VGPR pressure)
        }
    }

    // ---- epilogue: bias + store (C/D map: lanes 0-15 -> M=v, lanes 16-31 -> M=8+v, N=lane%16) ----
    const int mh = lane >> 4;
    #pragma unroll
    for (int tn = 0; tn < 4; ++tn) {
        int n = bn * BN + waveN * 64 + tn * 16 + lm;
        float bv = bias[n];
        #pragma unroll
        for (int tm = 0; tm < 4; ++tm) {
            int mbase = bm * BM + waveM * 64 + tm * 16 + mh * 8;
            #pragma unroll
            for (int v = 0; v < 8; ++v) {
                float r = acc[tm][tn][v] + bv;
                size_t idx = (size_t)(mbase + v) * H_DIM + n;
                if (F32OUT) ((float*)outp)[idx]  = r;
                else        ((__bf16*)outp)[idx] = (__bf16)r;
            }
        }
    }
}

extern "C" void kernel_launch(void* const* d_in, const int* in_sizes, int n_in,
                              void* d_out, int out_size, void* d_ws, size_t ws_size,
                              hipStream_t stream) {
    const float* x     = (const float*)d_in[0];
    const float* w     = (const float*)d_in[1];
    const float* bias  = (const float*)d_in[2];
    const int*   perms = (const int*)d_in[3];

    const size_t xElems = (size_t)B_DIM * S_DIM * H_DIM;       // 67,108,864
    const size_t wElems = (size_t)L_DIM * H_DIM * H_DIM;       // 16,777,216

    char* ws = (char*)d_ws;
    __bf16* xbf0 = (__bf16*)ws;
    __bf16* xbf1 = (__bf16*)(ws + xElems * 2);
    __bf16* wbf  = (__bf16*)(ws + xElems * 4);

    cvt_f32_to_bf16<<<(unsigned)(wElems / 8 / 256), 256, 0, stream>>>(w, wbf, (int)wElems);
    cvt_f32_to_bf16<<<(unsigned)(xElems / 8 / 256), 256, 0, stream>>>(x, xbf0, (int)xElems);

    dim3 grid(H_DIM / BN, (B_DIM * S_DIM) / BM);               // 8 x 256
    const size_t HH = (size_t)H_DIM * H_DIM;

    gemm_bf16_wmma<false><<<grid, 256, 0, stream>>>(xbf0, wbf + 0 * HH, bias + 0 * H_DIM,
                                                    nullptr,           xbf1);
    gemm_bf16_wmma<false><<<grid, 256, 0, stream>>>(xbf1, wbf + 1 * HH, bias + 1 * H_DIM,
                                                    perms + 0 * S_DIM, xbf0);
    gemm_bf16_wmma<false><<<grid, 256, 0, stream>>>(xbf0, wbf + 2 * HH, bias + 2 * H_DIM,
                                                    perms + 1 * S_DIM, xbf1);
    gemm_bf16_wmma<true ><<<grid, 256, 0, stream>>>(xbf1, wbf + 3 * HH, bias + 3 * H_DIM,
                                                    perms + 2 * S_DIM, d_out);
}
